// WindowedMultiHeadAttention_48868137893996
// MI455X (gfx1250) — matrix operand order repaired and checked by reading, not compiled, not run on hardware
//
#include <hip/hip_runtime.h>
#include <hip/hip_bf16.h>
#include <cstdint>

typedef __bf16 bf16;
typedef bf16  bf16x16 __attribute__((ext_vector_type(16)));
typedef bf16  bf16x8  __attribute__((ext_vector_type(8)));
typedef float v8f     __attribute__((ext_vector_type(8)));

#define DEV __device__ __forceinline__

constexpr int WSZ = 7;
constexpr int NT  = WSZ * WSZ;     // 49 tokens per window
constexpr int NP  = 64;            // padded token count
constexpr int C   = 128;
constexpr int H   = 4;
constexpr int HD  = C / H;         // 32
constexpr int NW  = 64;
constexpr int B   = 4096;
constexpr int MROWS = B * NT;      // 200704 (multiple of 16)
constexpr int MTILES = MROWS / 16; // 12544
constexpr int NN  = NT * NT;       // 2401
constexpr float SCALE = 0.17677669529663687f; // HD^-0.5
constexpr float NEGINF = -1e30f;

// packed-weight sizes (bf16 elems): per ntile: 4 ksteps * 32 lanes * 16 elems
constexpr int WQKV_NTILES = 24;    // 384 cols
constexpr int WPROJ_NTILES = 8;    // 128 cols
constexpr int WQKV_ELEMS  = WQKV_NTILES * 2048;   // 49152
constexpr int WPROJ_ELEMS = WPROJ_NTILES * 2048;  // 16384

union BF16x16 { bf16x16 v; bf16x8 h[2]; };

DEV v8f wmma_bf16(bf16x16 a, bf16x16 b, v8f c) {
    return __builtin_amdgcn_wmma_f32_16x16x32_bf16(
        false, a, false, b, (short)0, c, false, false);
}

DEV void zero16(BF16x16& t) {
#pragma unroll
    for (int i = 0; i < 16; ++i) t.v[i] = (bf16)0.0f;
}

// ---------------------------------------------------------------------------
// Prep A: pack [q_w | kv_w] and proj_w into bf16 WMMA-B-operand layout.
// Element (ntile, ks, lane, e) holds W[k = ks*32 + (e/8)*16 + (lane/16)*8 + e%8]
//                                     [col = ntile*16 + lane%16].
// q columns are pre-scaled by HD^-0.5.
// ---------------------------------------------------------------------------
__global__ void pack_w_kernel(const float* __restrict__ q_w,
                              const float* __restrict__ kv_w,
                              const float* __restrict__ proj_w,
                              bf16* __restrict__ wqkv,
                              bf16* __restrict__ wproj) {
    int t = blockIdx.x * blockDim.x + threadIdx.x;
    if (t < WQKV_ELEMS) {
        int e    = t & 15;
        int lane = (t >> 4) & 31;
        int ks   = (t >> 9) & 3;
        int nt   = t >> 11;
        int k    = ks * 32 + (e >> 3) * 16 + (lane >> 4) * 8 + (e & 7);
        int col  = nt * 16 + (lane & 15);
        float val;
        if (col < C) val = q_w[(size_t)k * C + col] * SCALE;
        else         val = kv_w[(size_t)k * (2 * C) + (col - C)];
        wqkv[t] = (bf16)val;
    } else if (t < WQKV_ELEMS + WPROJ_ELEMS) {
        int u = t - WQKV_ELEMS;
        int e    = u & 15;
        int lane = (u >> 4) & 31;
        int ks   = (u >> 9) & 3;
        int nt   = u >> 11;
        int k    = ks * 32 + (e >> 3) * 16 + (lane >> 4) * 8 + (e & 7);
        int col  = nt * 16 + (lane & 15);
        wproj[u] = (bf16)proj_w[(size_t)k * C + col];
    }
}

// ---------------------------------------------------------------------------
// Prep B: padded bias+mask tensor bm64[w][h][64][64] (fp32).
// Interior: bias_table[rel_idx[i][j]][h] + mask[w][i][j]; padding: -1e30.
// Fed directly as the C operand of the S = q k^T WMMA.
// ---------------------------------------------------------------------------
__global__ void bm64_kernel(const float* __restrict__ mask,
                            const float* __restrict__ bias_table,
                            const int*   __restrict__ rel_idx,
                            float* __restrict__ bm64) {
    int t = blockIdx.x * blockDim.x + threadIdx.x;
    int total = NW * H * NP * NP;
    if (t >= total) return;
    int j  = t & 63;
    int i  = (t >> 6) & 63;
    int wh = t >> 12;
    int h  = wh & (H - 1);
    int w  = wh / H;
    float val = NEGINF;
    if (i < NT && j < NT)
        val = bias_table[rel_idx[i * NT + j] * H + h] + mask[(size_t)w * NN + i * NT + j];
    bm64[t] = val;
}

// ---------------------------------------------------------------------------
// Prep C: zero the token-padding region of vt[b][h][d][49..63].
// ---------------------------------------------------------------------------
__global__ void vpad_kernel(bf16* __restrict__ vt) {
    int t = blockIdx.x * blockDim.x + threadIdx.x;
    int total = B * H * HD * (NP - NT);
    if (t >= total) return;
    int i   = t % (NP - NT);
    int bhd = t / (NP - NT);
    vt[(size_t)bhd * NP + NT + i] = (bf16)0.0f;
}

// ---------------------------------------------------------------------------
// Kernel 1: fused QKV projection (bf16 WMMA, f32 accumulate).
// qs/kk stored [B][H][NT][HD]; v stored transposed+padded vt[B][H][HD][64].
// ---------------------------------------------------------------------------
__global__ __launch_bounds__(256) void qkv_kernel(
    const float* __restrict__ query, const float* __restrict__ source,
    const bf16* __restrict__ wqkv,
    const float* __restrict__ q_b,   const float* __restrict__ kv_b,
    bf16* __restrict__ qs, bf16* __restrict__ kk, bf16* __restrict__ vt) {

    const int wave = threadIdx.x >> 5;
    const int lane = threadIdx.x & 31;
    const int half = lane >> 4;
    const int ln   = lane & 15;

    const int mtile = blockIdx.x;                 // 0..12543
    const int ntile = blockIdx.y * 8 + wave;      // 0..23 (384 cols)
    const int colg  = ntile * 16;

    const float* X = (colg < C) ? query : source;

    v8f acc = {0.f, 0.f, 0.f, 0.f, 0.f, 0.f, 0.f, 0.f};

#pragma unroll
    for (int ks = 0; ks < 4; ++ks) {              // K = 128 in steps of 32
        // A tile (16x32 bf16): row = lane%16, K = (e/8)*16 + half*8 + e%8
        const float* xr = X + (size_t)(mtile * 16 + ln) * C + ks * 32;
        BF16x16 a;
        const int k0 = half * 8;
#pragma unroll
        for (int e = 0; e < 8; ++e) a.v[e]     = (bf16)xr[k0 + e];
#pragma unroll
        for (int e = 0; e < 8; ++e) a.v[8 + e] = (bf16)xr[k0 + 16 + e];

        // B tile: packed, contiguous 16 bf16 per lane (two b128 loads)
        const bf16* wp = wqkv + ((size_t)(ntile * 4 + ks) * 32 + lane) * 16;
        BF16x16 bm;
        bm.h[0] = *reinterpret_cast<const bf16x8*>(wp);
        bm.h[1] = *reinterpret_cast<const bf16x8*>(wp + 8);

        acc = wmma_bf16(a.v, bm.v, acc);
    }

    // Epilogue: C/D layout row = mtile*16 + r + half*8, col = colg + ln
#pragma unroll
    for (int r = 0; r < 8; ++r) {
        int row = mtile * 16 + half * 8 + r;
        int b   = row / NT;
        int n   = row % NT;
        float val = acc[r];
        if (colg < C) {
            int c = colg + ln;
            int hh = c >> 5, d = c & 31;
            val += q_b[c] * SCALE;          // weights pre-scaled; scale bias here
            qs[(((size_t)b * H + hh) * NT + n) * HD + d] = (bf16)val;
        } else if (colg < 2 * C) {
            int c = colg - C + ln;
            int hh = c >> 5, d = c & 31;
            val += kv_b[c];
            kk[(((size_t)b * H + hh) * NT + n) * HD + d] = (bf16)val;
        } else {
            int c = colg - 2 * C + ln;
            int hh = c >> 5, d = c & 31;
            val += kv_b[C + c];
            vt[(((size_t)b * H + hh) * HD + d) * NP + n] = (bf16)val;  // transposed
        }
    }
}

// ---------------------------------------------------------------------------
// Kernel 2: windowed attention, one wave per (b, h).
// S = q k^T with bm64 as the WMMA C operand; row softmax via 16-lane
// shuffles; P through LDS (C/D layout -> A layout); O = P v.
// ---------------------------------------------------------------------------
__global__ __launch_bounds__(32) void attn_kernel(
    const bf16* __restrict__ qs, const bf16* __restrict__ kk,
    const bf16* __restrict__ vt, const float* __restrict__ bm64,
    bf16* __restrict__ xb) {

    const int bh = blockIdx.x;
    const int b  = bh / H;
    const int h  = bh % H;
    const int lane = threadIdx.x;
    const int half = lane >> 4;
    const int ln   = lane & 15;

    const bf16* qptr = qs + ((size_t)b * H + h) * NT * HD;
    const bf16* kptr = kk + ((size_t)b * H + h) * NT * HD;
    const bf16* vptr = vt + ((size_t)b * H + h) * HD * NP;
    const float* bp  = bm64 + ((size_t)((b % NW) * H + h)) * NP * NP;

    // ---- q as A tiles (rows = tokens, padded 49 -> 64) ----
    BF16x16 qa[4];
#pragma unroll
    for (int mt = 0; mt < 4; ++mt) {
        int row = mt * 16 + ln;
        if (row < NT) {
            const bf16* qr = qptr + (size_t)row * HD + half * 16;
            qa[mt].h[0] = *reinterpret_cast<const bf16x8*>(qr);
            qa[mt].h[1] = *reinterpret_cast<const bf16x8*>(qr + 8);
        } else {
            zero16(qa[mt]);
        }
    }

    // ---- k^T as B tiles: B(K=d, N=token), 16 contiguous d per lane ----
    BF16x16 kb[4];
#pragma unroll
    for (int nt = 0; nt < 4; ++nt) {
        int tok = nt * 16 + ln;
        if (tok < NT) {
            const bf16* kr = kptr + (size_t)tok * HD + half * 16;
            kb[nt].h[0] = *reinterpret_cast<const bf16x8*>(kr);
            kb[nt].h[1] = *reinterpret_cast<const bf16x8*>(kr + 8);
        } else {
            zero16(kb[nt]);
        }
    }

    // ---- S = q k^T + (bias+mask) via C operand ----
    v8f s[4][4];
#pragma unroll
    for (int mt = 0; mt < 4; ++mt)
#pragma unroll
        for (int nt = 0; nt < 4; ++nt) {
            v8f cini;
#pragma unroll
            for (int r = 0; r < 8; ++r)
                cini[r] = bp[(mt * 16 + half * 8 + r) * NP + nt * 16 + ln];
            s[mt][nt] = wmma_bf16(qa[mt].v, kb[nt].v, cini);
        }

    // ---- row softmax (rows live in 16-lane halves) ----
    __shared__ __align__(16) bf16 Pb[NP * NP];

#pragma unroll
    for (int mt = 0; mt < 4; ++mt) {
#pragma unroll
        for (int r = 0; r < 8; ++r) {
            int row = mt * 16 + half * 8 + r;
            float vals[4];
#pragma unroll
            for (int nt = 0; nt < 4; ++nt) vals[nt] = s[mt][nt][r];
            float mx = fmaxf(fmaxf(vals[0], vals[1]), fmaxf(vals[2], vals[3]));
            mx = fmaxf(mx, __shfl_xor(mx, 1, 32));
            mx = fmaxf(mx, __shfl_xor(mx, 2, 32));
            mx = fmaxf(mx, __shfl_xor(mx, 4, 32));
            mx = fmaxf(mx, __shfl_xor(mx, 8, 32));
            float ssum = 0.f;
#pragma unroll
            for (int nt = 0; nt < 4; ++nt) {
                vals[nt] = __expf(vals[nt] - mx);
                ssum += vals[nt];
            }
            ssum += __shfl_xor(ssum, 1, 32);
            ssum += __shfl_xor(ssum, 2, 32);
            ssum += __shfl_xor(ssum, 4, 32);
            ssum += __shfl_xor(ssum, 8, 32);
            float inv = 1.0f / (ssum + 1e-20f);
#pragma unroll
            for (int nt = 0; nt < 4; ++nt)
                Pb[row * NP + nt * 16 + ln] = (bf16)(vals[nt] * inv);
        }
    }
    __syncthreads();

    // ---- v as B tiles from transposed+padded vt: contiguous, unguarded ----
    BF16x16 vb[2][2];
#pragma unroll
    for (int kt = 0; kt < 2; ++kt)
#pragma unroll
        for (int nt = 0; nt < 2; ++nt) {
            const bf16* vr = vptr + (size_t)(nt * 16 + ln) * NP + kt * 32 + half * 16;
            vb[kt][nt].h[0] = *reinterpret_cast<const bf16x8*>(vr);
            vb[kt][nt].h[1] = *reinterpret_cast<const bf16x8*>(vr + 8);
        }

    // ---- O = P v ----
#pragma unroll
    for (int mt = 0; mt < 4; ++mt) {
        BF16x16 pa[2];
#pragma unroll
        for (int kt = 0; kt < 2; ++kt) {
            int row = mt * 16 + ln;
            const bf16* pr = Pb + row * NP + kt * 32 + half * 16;
            pa[kt].h[0] = *reinterpret_cast<const bf16x8*>(pr);
            pa[kt].h[1] = *reinterpret_cast<const bf16x8*>(pr + 8);
        }
#pragma unroll
        for (int nt = 0; nt < 2; ++nt) {
            v8f o = {0.f, 0.f, 0.f, 0.f, 0.f, 0.f, 0.f, 0.f};
            o = wmma_bf16(pa[0].v, vb[0][nt].v, o);
            o = wmma_bf16(pa[1].v, vb[1][nt].v, o);
#pragma unroll
            for (int r = 0; r < 8; ++r) {
                int tok = mt * 16 + half * 8 + r;
                if (tok < NT)
                    xb[((size_t)b * NT + tok) * C + h * HD + nt * 16 + ln] = (bf16)o[r];
            }
        }
    }
}

// ---------------------------------------------------------------------------
// Kernel 3: output projection. xb[200704x128] bf16 @ packed proj_w -> f32.
// ---------------------------------------------------------------------------
__global__ __launch_bounds__(256) void proj_kernel(
    const bf16* __restrict__ xb, const bf16* __restrict__ wproj,
    const float* __restrict__ proj_b, float* __restrict__ out) {

    const int wave = threadIdx.x >> 5;
    const int lane = threadIdx.x & 31;
    const int half = lane >> 4;
    const int ln   = lane & 15;

    const int mtile = blockIdx.x;
    const int col   = wave * 16 + ln;    // 8 waves cover 128 cols

    v8f acc = {0.f, 0.f, 0.f, 0.f, 0.f, 0.f, 0.f, 0.f};

#pragma unroll
    for (int ks = 0; ks < 4; ++ks) {
        const bf16* xr = xb + (size_t)(mtile * 16 + ln) * C + ks * 32 + half * 8;
        BF16x16 a;
        a.h[0] = *reinterpret_cast<const bf16x8*>(xr);
        a.h[1] = *reinterpret_cast<const bf16x8*>(xr + 16);

        const bf16* wp = wproj + ((size_t)(wave * 4 + ks) * 32 + lane) * 16;
        BF16x16 bm;
        bm.h[0] = *reinterpret_cast<const bf16x8*>(wp);
        bm.h[1] = *reinterpret_cast<const bf16x8*>(wp + 8);

        acc = wmma_bf16(a.v, bm.v, acc);
    }

#pragma unroll
    for (int r = 0; r < 8; ++r) {
        int row = mtile * 16 + half * 8 + r;
        out[(size_t)row * C + col] = acc[r] + proj_b[col];
    }
}

// ---------------------------------------------------------------------------
extern "C" void kernel_launch(void* const* d_in, const int* in_sizes, int n_in,
                              void* d_out, int out_size, void* d_ws, size_t ws_size,
                              hipStream_t stream) {
    const float* query      = (const float*)d_in[0];
    const float* source     = (const float*)d_in[1];
    const float* mask       = (const float*)d_in[2];
    const float* q_w        = (const float*)d_in[3];
    const float* q_b        = (const float*)d_in[4];
    const float* kv_w       = (const float*)d_in[5];
    const float* kv_b       = (const float*)d_in[6];
    const float* proj_w     = (const float*)d_in[7];
    const float* proj_b     = (const float*)d_in[8];
    const float* bias_table = (const float*)d_in[9];
    const int*   rel_idx    = (const int*)d_in[10];
    float* out = (float*)d_out;

    // Workspace partition (all slabs 256B-aligned by construction)
    const size_t QE  = (size_t)B * H * NT * HD;   // 25,690,112 bf16 elems
    const size_t VTE = (size_t)B * H * HD * NP;   // 33,554,432 bf16 elems
    bf16* qs    = (bf16*)d_ws;
    bf16* kk    = qs + QE;
    bf16* vt    = kk + QE;                        // [B][H][HD][64]
    bf16* xb    = vt + VTE;                       // [B][NT][C]
    bf16* wqkv  = xb + QE;                        // packed
    bf16* wproj = wqkv + WQKV_ELEMS;
    float* bm64 = (float*)(wproj + WPROJ_ELEMS);  // [NW][H][64][64]

    const int packTotal = WQKV_ELEMS + WPROJ_ELEMS;
    pack_w_kernel<<<(packTotal + 255) / 256, 256, 0, stream>>>(q_w, kv_w, proj_w,
                                                               wqkv, wproj);

    const int bmTotal = NW * H * NP * NP;
    bm64_kernel<<<(bmTotal + 255) / 256, 256, 0, stream>>>(mask, bias_table,
                                                           rel_idx, bm64);

    const int padTotal = B * H * HD * (NP - NT);
    vpad_kernel<<<(padTotal + 255) / 256, 256, 0, stream>>>(vt);

    qkv_kernel<<<dim3(MTILES, 3), 256, 0, stream>>>(query, source, wqkv,
                                                    q_b, kv_b, qs, kk, vt);

    attn_kernel<<<B * H, 32, 0, stream>>>(qs, kk, vt, bm64, xb);

    proj_kernel<<<MTILES, 256, 0, stream>>>(xb, wproj, proj_b, out);

    (void)in_sizes; (void)n_in; (void)out_size; (void)ws_size;
}
